// SparseTransformer_62672162783471
// MI455X (gfx1250) — matrix operand order repaired and checked by reading, not compiled, not run on hardware
//
#include <hip/hip_runtime.h>
#include <hip/hip_bf16.h>
#include <cstdint>
#include <cstddef>

typedef __bf16 bf16;
typedef __attribute__((ext_vector_type(8)))  bf16  v8bf;
typedef __attribute__((ext_vector_type(16))) bf16  v16bf;
typedef __attribute__((ext_vector_type(8)))  float v8f;
typedef __attribute__((ext_vector_type(4)))  int   v4i;

constexpr int BB   = 128;
constexpr int SEQ  = 53;
constexpr int DIM  = 1024;
constexpr int NH   = 16;
constexpr int NL   = 6;
constexpr int FF   = 4096;
constexpr int DK   = DIM / NH;      // 64
constexpr int MTOK = BB * SEQ;      // 6784 = 53 * 128

// ---------------------------------------------------------------------------
// CDNA5 async global->LDS copy (ASYNCcnt path), with register-pipelined
// fallback if the builtin is not available on this toolchain.
// The builtin expects int4 pointers: (global int4*, lds int4*, imm off, imm cpol).
// ---------------------------------------------------------------------------
#if __has_builtin(__builtin_amdgcn_global_load_async_to_lds_b128)
#define USE_ASYNC_LDS 1
typedef __attribute__((address_space(1))) v4i* gas1_v4i;
typedef __attribute__((address_space(3))) v4i* gas3_v4i;
__device__ __forceinline__ void async_cp16(const bf16* g, bf16* l) {
  __builtin_amdgcn_global_load_async_to_lds_b128(
      (gas1_v4i)(void*)g, (gas3_v4i)(void*)l, 0, 0);
}
#else
#define USE_ASYNC_LDS 0
#endif

__device__ __forceinline__ void wait_async_lds() {
#if USE_ASYNC_LDS
#if __has_builtin(__builtin_amdgcn_s_wait_asynccnt)
  __builtin_amdgcn_s_wait_asynccnt(0);
#else
  asm volatile("s_wait_asynccnt 0" ::: "memory");
#endif
#endif
}

// ---------------------------------------------------------------------------
// Embedding + token overrides + per-position batchnorm.  One block per token.
// ---------------------------------------------------------------------------
__global__ __launch_bounds__(256) void embed_bn(
    const int* __restrict__ x, const float* __restrict__ in_w, const float* __restrict__ in_b,
    const float* __restrict__ sos_tok, const float* __restrict__ sep_tok,
    const float* __restrict__ act_emb,
    const float* __restrict__ bn_g, const float* __restrict__ bn_b,
    const float* __restrict__ bn_mean, const float* __restrict__ bn_var,
    float* __restrict__ hF, bf16* __restrict__ hB)
{
  __shared__ int sActIdx, sAction;
  const int bs = blockIdx.x;
  const int b = bs / SEQ, s = bs % SEQ;
  if (threadIdx.x == 0) {
    int sepIdx = 0;
    for (int j = 0; j < SEQ; ++j) {
      const int* xr = x + (size_t)(b * SEQ + j) * 3;
      if (xr[0] == 100 && xr[1] == 0 && xr[2] == 2) { sepIdx = j; break; }
    }
    int ai = sepIdx - 1;
    if (ai < 0) ai = 0;
    sActIdx = ai;
    sAction = x[(size_t)(b * SEQ + ai) * 3 + 2];
  }
  __syncthreads();
  const int* xr = x + (size_t)(b * SEQ + s) * 3;
  const int x0 = xr[0], x1 = xr[1], x2 = xr[2];
  const bool isSos = (x0 == 100 && x1 == 0 && x2 == 1);
  const bool isSep = (x0 == 100 && x1 == 0 && x2 == 2);
  const bool isAct = (s == sActIdx);
  const float f0 = (float)x0, f1 = (float)x1, f2 = (float)x2;
  const float rstd = rsqrtf(bn_var[s] + 1e-5f);
  const float mu = bn_mean[s], gg = bn_g[s], bb = bn_b[s];
  const int action = sAction;
  for (int d = threadIdx.x; d < DIM; d += 256) {
    float v = f0 * in_w[d] + f1 * in_w[DIM + d] + f2 * in_w[2 * DIM + d] + in_b[d];
    if (isSos) v = sos_tok[d];
    if (isSep) v = sep_tok[d];
    if (isAct) v = act_emb[(size_t)action * DIM + d];
    v = (v - mu) * rstd * gg + bb;
    const size_t idx = (size_t)bs * DIM + d;
    hF[idx] = v;
    hB[idx] = (bf16)v;
  }
}

// ---------------------------------------------------------------------------
// f32 (K x N) weight  ->  bf16 (N x K) transposed copy, 32x32 LDS tiles.
// ---------------------------------------------------------------------------
__global__ __launch_bounds__(256) void transpose_to_bf16(
    const float* __restrict__ W, bf16* __restrict__ WT, int K, int N)
{
  __shared__ float tile[32][33];
  const int nb = blockIdx.x * 32, kb = blockIdx.y * 32;
  const int tx = threadIdx.x, ty = threadIdx.y;   // 32 x 8
#pragma unroll
  for (int i = 0; i < 32; i += 8)
    tile[ty + i][tx] = W[(size_t)(kb + ty + i) * N + nb + tx];
  __syncthreads();
#pragma unroll
  for (int i = 0; i < 32; i += 8)
    WT[(size_t)(nb + ty + i) * K + kb + tx] = (bf16)tile[tx][ty + i];
}

// ---------------------------------------------------------------------------
// WMMA GEMM:  C[M,N] = A[M,K] * BT[N,K]^T + bias, compile-time epilogue mode.
// 256 threads = 8 waves; block tile 128x128, BK = 32, double-buffered LDS.
// Wave (wm in 0..3, wn in 0..1) owns a 32x64 sub-tile = 2x4 wmma tiles.
// Staging uses CDNA5 async global->LDS copies (ASYNCcnt) when available.
// ---------------------------------------------------------------------------
constexpr int LDS_STRIDE = 40;           // elems; 80B rows keep 16B alignment
constexpr int TILE_ELEMS = 128 * LDS_STRIDE;

template<bool RELU, bool STORE_F32, bool STORE_BF16>
__global__ __launch_bounds__(256) void wmma_gemm(
    const bf16* __restrict__ A, const bf16* __restrict__ BT,
    const float* __restrict__ bias,
    float* __restrict__ Cf, bf16* __restrict__ Ch,
    int M, int N, int K)
{
  __shared__ alignas(16) bf16 As[2][TILE_ELEMS];
  __shared__ alignas(16) bf16 Bs[2][TILE_ELEMS];

  const int tid  = threadIdx.x;
  const int lane = tid & 31;
  const int wave = tid >> 5;
  const int half = lane >> 4;      // 0: lanes 0-15, 1: lanes 16-31
  const int l16  = lane & 15;
  const int wm   = wave & 3;       // M sub-tile
  const int wn   = wave >> 2;      // N sub-tile
  const int mBase = blockIdx.y * 128;
  const int nBase = blockIdx.x * 128;

  v8f acc[2][4];
#pragma unroll
  for (int tm = 0; tm < 2; ++tm)
#pragma unroll
    for (int tn = 0; tn < 4; ++tn)
#pragma unroll
      for (int r = 0; r < 8; ++r) acc[tm][tn][r] = 0.0f;

  // staging map: thread t owns 16 bf16 of row (t>>1), segment (t&1)*16
  const int ldRow = tid >> 1;
  const int ldSeg = (tid & 1) * 16;
  const bf16* gA = A  + (size_t)(mBase + ldRow) * K + ldSeg;
  const bf16* gB = BT + (size_t)(nBase + ldRow) * K + ldSeg;
  const int sOff = ldRow * LDS_STRIDE + ldSeg;

  // ---- preload tile 0 into buffer 0 ----
#if USE_ASYNC_LDS
  async_cp16(gA,     &As[0][sOff]);
  async_cp16(gA + 8, &As[0][sOff + 8]);
  async_cp16(gB,     &Bs[0][sOff]);
  async_cp16(gB + 8, &Bs[0][sOff + 8]);
  wait_async_lds();
#else
  {
    v8bf ra0 = *(const v8bf*)(gA);
    v8bf ra1 = *(const v8bf*)(gA + 8);
    v8bf rb0 = *(const v8bf*)(gB);
    v8bf rb1 = *(const v8bf*)(gB + 8);
    *(v8bf*)&As[0][sOff]     = ra0;
    *(v8bf*)&As[0][sOff + 8] = ra1;
    *(v8bf*)&Bs[0][sOff]     = rb0;
    *(v8bf*)&Bs[0][sOff + 8] = rb1;
  }
#endif
  __syncthreads();

  int buf = 0;
  for (int k0 = 0; k0 < K; k0 += 32) {
    const bool hasNext = (k0 + 32) < K;
    const int nxt = buf ^ 1;

    // ---- issue next tile's copies before touching this tile ----
#if USE_ASYNC_LDS
    if (hasNext) {
      async_cp16(gA + k0 + 32,     &As[nxt][sOff]);
      async_cp16(gA + k0 + 40,     &As[nxt][sOff + 8]);
      async_cp16(gB + k0 + 32,     &Bs[nxt][sOff]);
      async_cp16(gB + k0 + 40,     &Bs[nxt][sOff + 8]);
    }
#else
    v8bf ra0, ra1, rb0, rb1;
    if (hasNext) {
      ra0 = *(const v8bf*)(gA + k0 + 32);
      ra1 = *(const v8bf*)(gA + k0 + 40);
      rb0 = *(const v8bf*)(gB + k0 + 32);
      rb1 = *(const v8bf*)(gB + k0 + 40);
    }
#endif

    // ---- fragments from LDS[buf] ----
    // A fragment (16x32 bf16): lane holds K in [half*8, +8) and [16+half*8, +8)
    v16bf aF[2];
#pragma unroll
    for (int tm = 0; tm < 2; ++tm) {
      const bf16* p = &As[buf][(wm * 32 + tm * 16 + l16) * LDS_STRIDE];
      v8bf lo = *(const v8bf*)(p + half * 8);
      v8bf hi = *(const v8bf*)(p + 16 + half * 8);
#pragma unroll
      for (int i = 0; i < 8; ++i) { aF[tm][i] = lo[i]; aF[tm][i + 8] = hi[i]; }
    }
    // B fragment (32x16 bf16): lane (col = l16) holds K in [half*8, +8) and [16+half*8, +8)
    v16bf bF[4];
#pragma unroll
    for (int tn = 0; tn < 4; ++tn) {
      const bf16* p = &Bs[buf][(wn * 64 + tn * 16 + l16) * LDS_STRIDE];
      v8bf lo = *(const v8bf*)(p + half * 8);
      v8bf hi = *(const v8bf*)(p + 16 + half * 8);
#pragma unroll
      for (int i = 0; i < 8; ++i) { bF[tn][i] = lo[i]; bF[tn][i + 8] = hi[i]; }
    }
#pragma unroll
    for (int tm = 0; tm < 2; ++tm)
#pragma unroll
      for (int tn = 0; tn < 4; ++tn)
        acc[tm][tn] = __builtin_amdgcn_wmma_f32_16x16x32_bf16(
            false, aF[tm], false, bF[tn], (short)0, acc[tm][tn], false, false);

    // ---- complete next tile's staging, single barrier per K-step ----
#if USE_ASYNC_LDS
    if (hasNext) wait_async_lds();
#else
    if (hasNext) {
      *(v8bf*)&As[nxt][sOff]     = ra0;
      *(v8bf*)&As[nxt][sOff + 8] = ra1;
      *(v8bf*)&Bs[nxt][sOff]     = rb0;
      *(v8bf*)&Bs[nxt][sOff + 8] = rb1;
    }
#endif
    __syncthreads();
    buf = nxt;
  }

  // epilogue: C lane map — lanes 0-15: (M=r, N=lane), lanes 16-31: (M=8+r, N=lane-16)
#pragma unroll
  for (int tm = 0; tm < 2; ++tm) {
#pragma unroll
    for (int tn = 0; tn < 4; ++tn) {
      const int n = nBase + wn * 64 + tn * 16 + l16;
      const float bv = bias[n];
#pragma unroll
      for (int r = 0; r < 8; ++r) {
        const int m = mBase + wm * 32 + tm * 16 + half * 8 + r;
        float v = acc[tm][tn][r] + bv;
        if (RELU) v = v > 0.0f ? v : 0.0f;
        const size_t idx = (size_t)m * N + n;
        if (STORE_F32)  Cf[idx] = v;
        if (STORE_BF16) Ch[idx] = (bf16)v;
      }
    }
  }
}

// ---------------------------------------------------------------------------
// Sparse attention, one block per (batch, head).  Reproduces the reference
// exactly: inverted pad mask (keep score where key IS pad), top-5 threshold,
// softmax over zero-filled sparse scores (exp(0)=1 terms included).
// ---------------------------------------------------------------------------
__global__ __launch_bounds__(64) void attn_sparse(
    const bf16* __restrict__ qB, const bf16* __restrict__ kB, const bf16* __restrict__ vB,
    const int* __restrict__ x, bf16* __restrict__ oB)
{
  __shared__ float ks[SEQ][DK + 1];
  __shared__ float vs[SEQ][DK + 1];
  __shared__ float scs[64][SEQ + 4];   // padded stride 57 to dodge bank conflicts
  __shared__ int smask[SEQ];

  const int bh = blockIdx.x;
  const int b = bh / NH, hh = bh % NH;
  const int t = threadIdx.x;

  for (int e = t; e < SEQ * DK; e += 64) {
    const int j = e >> 6, d = e & 63;
    const size_t idx = (size_t)(b * SEQ + j) * DIM + hh * DK + d;
    ks[j][d] = (float)kB[idx];
    vs[j][d] = (float)vB[idx];
  }
  if (t < SEQ) {
    const int* xr = x + (size_t)(b * SEQ + t) * 3;
    smask[t] = (xr[0] == 100 && xr[1] == 0 && xr[2] == 0) ? 1 : 0;
  }
  __syncthreads();

  if (t < SEQ) {
    float qreg[DK];
    const size_t qidx = (size_t)(b * SEQ + t) * DIM + hh * DK;
#pragma unroll
    for (int d = 0; d < DK; ++d) qreg[d] = (float)qB[qidx + d];

    const float scale = 0.125f;        // 1/sqrt(64)
    for (int j = 0; j < SEQ; ++j) {
      float s = 0.0f;
#pragma unroll
      for (int d = 0; d < DK; ++d) s += qreg[d] * ks[j][d];
      scs[t][j] = smask[j] ? s * scale : -1.0e9f;   // reference keeps score at PAD keys
    }
    // top-5 threshold
    float t5[5] = { -3.4e38f, -3.4e38f, -3.4e38f, -3.4e38f, -3.4e38f };
    for (int j = 0; j < SEQ; ++j) {
      float sv = scs[t][j];
#pragma unroll
      for (int i = 0; i < 5; ++i)
        if (sv > t5[i]) { const float tmp = t5[i]; t5[i] = sv; sv = tmp; }
    }
    const float thr = t5[4];
    // sparse_scores = scores * (scores >= thr); softmax over that (zeros included)
    float mx = -3.4e38f;
    for (int j = 0; j < SEQ; ++j) {
      const float sv = scs[t][j];
      const float val = (sv >= thr) ? sv : 0.0f;
      scs[t][j] = val;
      mx = val > mx ? val : mx;
    }
    float sum = 0.0f;
    for (int j = 0; j < SEQ; ++j) {
      const float e = __expf(scs[t][j] - mx);
      scs[t][j] = e;
      sum += e;
    }
    const float inv = 1.0f / sum;
    float accd[DK];
#pragma unroll
    for (int d = 0; d < DK; ++d) accd[d] = 0.0f;
    for (int j = 0; j < SEQ; ++j) {
      const float a = scs[t][j] * inv;
#pragma unroll
      for (int d = 0; d < DK; ++d) accd[d] += a * vs[j][d];
    }
    const size_t oidx = (size_t)(b * SEQ + t) * DIM + hh * DK;
#pragma unroll
    for (int d = 0; d < DK; ++d) oB[oidx + d] = (bf16)accd[d];
  }
}

// ---------------------------------------------------------------------------
// h = LayerNorm(resid + delta) * g + b ; writes f32 + bf16 copies (in-place
// safe: each thread re-writes exactly the elements it read before the syncs).
// ---------------------------------------------------------------------------
__global__ __launch_bounds__(256) void residual_ln(
    const float* resid, const float* __restrict__ delta,
    const float* __restrict__ g, const float* __restrict__ bta,
    float* hF, bf16* __restrict__ hB)
{
  __shared__ float rs[256], rq[256];
  const int row = blockIdx.x;
  const int t = threadIdx.x;
  float loc[4];
  float s = 0.0f, s2 = 0.0f;
#pragma unroll
  for (int i = 0; i < 4; ++i) {
    const int d = t + i * 256;
    const float v = resid[(size_t)row * DIM + d] + delta[(size_t)row * DIM + d];
    loc[i] = v; s += v; s2 += v * v;
  }
  rs[t] = s; rq[t] = s2;
  __syncthreads();
  for (int st = 128; st > 0; st >>= 1) {
    if (t < st) { rs[t] += rs[t + st]; rq[t] += rq[t + st]; }
    __syncthreads();
  }
  const float mean = rs[0] * (1.0f / DIM);
  const float var  = rq[0] * (1.0f / DIM) - mean * mean;
  const float rstd = rsqrtf(var + 1e-5f);
#pragma unroll
  for (int i = 0; i < 4; ++i) {
    const int d = t + i * 256;
    const float y = (loc[i] - mean) * rstd * g[d] + bta[d];
    hF[(size_t)row * DIM + d] = y;
    hB[(size_t)row * DIM + d] = (bf16)y;
  }
}

// ---------------------------------------------------------------------------
// out[b, :] = h[b, S-1, :] @ out_w (1024x3) + out_b
// ---------------------------------------------------------------------------
__global__ __launch_bounds__(128) void out_proj(
    const float* __restrict__ hF, const float* __restrict__ out_w,
    const float* __restrict__ out_b, float* __restrict__ out)
{
  __shared__ float r0[128], r1[128], r2[128];
  const int b = blockIdx.x, t = threadIdx.x;
  const float* row = hF + ((size_t)b * SEQ + (SEQ - 1)) * DIM;
  float a0 = 0.0f, a1 = 0.0f, a2 = 0.0f;
  for (int d = t; d < DIM; d += 128) {
    const float v = row[d];
    a0 += v * out_w[d * 3 + 0];
    a1 += v * out_w[d * 3 + 1];
    a2 += v * out_w[d * 3 + 2];
  }
  r0[t] = a0; r1[t] = a1; r2[t] = a2;
  __syncthreads();
  for (int st = 64; st > 0; st >>= 1) {
    if (t < st) { r0[t] += r0[t + st]; r1[t] += r1[t + st]; r2[t] += r2[t + st]; }
    __syncthreads();
  }
  if (t == 0) {
    out[b * 3 + 0] = r0[0] + out_b[0];
    out[b * 3 + 1] = r1[0] + out_b[1];
    out[b * 3 + 2] = r2[0] + out_b[2];
  }
}

// ---------------------------------------------------------------------------
extern "C" void kernel_launch(void* const* d_in, const int* in_sizes, int n_in,
                              void* d_out, int out_size, void* d_ws, size_t ws_size,
                              hipStream_t stream)
{
  (void)in_sizes; (void)n_in; (void)out_size; (void)ws_size;
  const int*   x       = (const int*)  d_in[0];
  const float* in_w    = (const float*)d_in[1];
  const float* in_b    = (const float*)d_in[2];
  const float* sos_tok = (const float*)d_in[3];
  const float* sep_tok = (const float*)d_in[4];
  const float* act_emb = (const float*)d_in[5];
  const float* bn_g    = (const float*)d_in[6];
  const float* bn_b    = (const float*)d_in[7];
  const float* bn_mean = (const float*)d_in[8];
  const float* bn_var  = (const float*)d_in[9];
  const float* Wq = (const float*)d_in[10]; const float* bq = (const float*)d_in[11];
  const float* Wk = (const float*)d_in[12]; const float* bk = (const float*)d_in[13];
  const float* Wv = (const float*)d_in[14]; const float* bv = (const float*)d_in[15];
  const float* Wo = (const float*)d_in[16]; const float* bo = (const float*)d_in[17];
  const float* ln1_g = (const float*)d_in[18]; const float* ln1_b = (const float*)d_in[19];
  const float* W1 = (const float*)d_in[20]; const float* b1 = (const float*)d_in[21];
  const float* W2 = (const float*)d_in[22]; const float* b2 = (const float*)d_in[23];
  const float* ln2_g = (const float*)d_in[24]; const float* ln2_b = (const float*)d_in[25];
  const float* out_w = (const float*)d_in[26]; const float* out_b = (const float*)d_in[27];

  char* ws = (char*)d_ws;
  size_t off = 0;
  auto take = [&](size_t bytes) -> char* {
    char* p = ws + off;
    off += (bytes + 255) & ~(size_t)255;
    return p;
  };
  float* hF   = (float*)take((size_t)MTOK * DIM * 4);
  bf16*  hB   = (bf16*) take((size_t)MTOK * DIM * 2);
  bf16*  qB   = (bf16*) take((size_t)MTOK * DIM * 2);
  bf16*  kB   = (bf16*) take((size_t)MTOK * DIM * 2);
  bf16*  vB   = (bf16*) take((size_t)MTOK * DIM * 2);
  bf16*  oB   = (bf16*) take((size_t)MTOK * DIM * 2);
  float* tmpF = (float*)take((size_t)MTOK * DIM * 4);
  bf16*  ffB  = (bf16*) take((size_t)MTOK * FF * 2);
  bf16*  wtB  = (bf16*) take((size_t)DIM * FF * 2);   // reused per-GEMM transposed weight

  embed_bn<<<MTOK, 256, 0, stream>>>(x, in_w, in_b, sos_tok, sep_tok, act_emb,
                                     bn_g, bn_b, bn_mean, bn_var, hF, hB);

  const dim3 tb(32, 8);
  const dim3 gD(DIM / 32, DIM / 32);         // D x D weight transpose
  const dim3 g1(FF / 32, DIM / 32);          // W1: K=D rows, N=FF cols
  const dim3 g2(DIM / 32, FF / 32);          // W2: K=FF rows, N=D cols
  const dim3 gemmD(DIM / 128, MTOK / 128);   // N = 1024
  const dim3 gemmF(FF / 128, MTOK / 128);    // N = 4096

  for (int l = 0; l < NL; ++l) {
    const size_t wOff = (size_t)l * DIM * DIM;
    // Q / K / V projections (bf16 outputs for the attention kernel)
    transpose_to_bf16<<<gD, tb, 0, stream>>>(Wq + wOff, wtB, DIM, DIM);
    wmma_gemm<false, false, true><<<gemmD, 256, 0, stream>>>(
        hB, wtB, bq + l * DIM, nullptr, qB, MTOK, DIM, DIM);
    transpose_to_bf16<<<gD, tb, 0, stream>>>(Wk + wOff, wtB, DIM, DIM);
    wmma_gemm<false, false, true><<<gemmD, 256, 0, stream>>>(
        hB, wtB, bk + l * DIM, nullptr, kB, MTOK, DIM, DIM);
    transpose_to_bf16<<<gD, tb, 0, stream>>>(Wv + wOff, wtB, DIM, DIM);
    wmma_gemm<false, false, true><<<gemmD, 256, 0, stream>>>(
        hB, wtB, bv + l * DIM, nullptr, vB, MTOK, DIM, DIM);
    // sparse attention
    attn_sparse<<<BB * NH, 64, 0, stream>>>(qB, kB, vB, x, oB);
    // output projection + residual + LN1
    transpose_to_bf16<<<gD, tb, 0, stream>>>(Wo + wOff, wtB, DIM, DIM);
    wmma_gemm<false, true, false><<<gemmD, 256, 0, stream>>>(
        oB, wtB, bo + l * DIM, tmpF, nullptr, MTOK, DIM, DIM);
    residual_ln<<<MTOK, 256, 0, stream>>>(hF, tmpF, ln1_g + l * DIM, ln1_b + l * DIM, hF, hB);
    // FFN: relu(h @ W1 + b1) @ W2 + b2, residual + LN2
    transpose_to_bf16<<<g1, tb, 0, stream>>>(W1 + (size_t)l * DIM * FF, wtB, DIM, FF);
    wmma_gemm<true, false, true><<<gemmF, 256, 0, stream>>>(
        hB, wtB, b1 + (size_t)l * FF, nullptr, ffB, MTOK, FF, DIM);
    transpose_to_bf16<<<g2, tb, 0, stream>>>(W2 + (size_t)l * FF * DIM, wtB, FF, DIM);
    wmma_gemm<false, true, false><<<gemmD, 256, 0, stream>>>(
        ffB, wtB, b2 + l * DIM, tmpF, nullptr, MTOK, DIM, FF);
    residual_ln<<<MTOK, 256, 0, stream>>>(hF, tmpF, ln2_g + l * DIM, ln2_b + l * DIM, hF, hB);
  }

  out_proj<<<BB, 128, 0, stream>>>(hF, out_w, out_b, (float*)d_out);
}
